// CausalHistoryModel_33011118637112
// MI455X (gfx1250) — matrix operand order repaired and checked by reading, not compiled, not run on hardware
//
#include <hip/hip_runtime.h>
#include <cstdint>
#include <cstddef>

// ---------------------------------------------------------------------------
// CDNA5 (gfx1250) implementation of CausalHistoryModel forward.
// Dense matmul stages use v_wmma_f32_16x16x32_bf16 (wave32 WMMA, f32 accum).
// Conv inputs are pre-converted once to pixel-major bf16 so WMMA fragments are
// contiguous 16B loads. Sparse attention uses deterministic top5+local lists.
// ---------------------------------------------------------------------------

typedef __attribute__((ext_vector_type(16))) __bf16 bf16x16;
typedef __attribute__((ext_vector_type(8)))  __bf16 bf16x8;
typedef __attribute__((ext_vector_type(8)))  float  f32x8;

__device__ __forceinline__ __bf16 f2bf(float f) {
  union { float f; unsigned u; } c; c.f = f;
  unsigned u = c.u;
  u += 0x7FFFu + ((u >> 16) & 1u);           // round-to-nearest-even
  unsigned short h = (unsigned short)(u >> 16);
  union { unsigned short s; __bf16 b; } o; o.s = h;
  return o.b;
}

__device__ __forceinline__ float wave_max(float v) {
  for (int o = 16; o > 0; o >>= 1) v = fmaxf(v, __shfl_xor(v, o, 32));
  return v;
}
__device__ __forceinline__ float wave_sum(float v) {
  for (int o = 16; o > 0; o >>= 1) v += __shfl_xor(v, o, 32);
  return v;
}

// -------------------------- weight f32 -> bf16 ------------------------------
__global__ void chm_cvt_bf16(const float* __restrict__ src, __bf16* __restrict__ dst, int n) {
  int i = blockIdx.x * 256 + threadIdx.x;
  if (i < n) dst[i] = f2bf(src[i]);
}

// ---- channel-major f32 (b,CinTot,P) -> pixel-major bf16 (b,P,Cin) ----------
// LDS 32x32 tile transpose; both global sides coalesced.
// grid (P/32, Cin/32, B), block 256.
__global__ void chm_cvt_tr_bf16(const float* __restrict__ in, __bf16* __restrict__ out,
                                int CinTot, int cofs, int Cin) {
  __shared__ __bf16 tile[32][33];
  const int t = threadIdx.x;
  const int p0 = blockIdx.x << 5;
  const int c0 = blockIdx.y << 5;
  const int b = blockIdx.z;
  const int px = t & 31, cg = t >> 5;
  for (int cc = cg; cc < 32; cc += 8)
    tile[cc][px] = f2bf(in[((size_t)b * CinTot + cofs + c0 + cc) * 65536 + p0 + px]);
  __syncthreads();
  const int c = t & 31, pg = t >> 5;
  for (int pp = pg; pp < 32; pp += 8)
    out[((size_t)b * 65536 + p0 + pp) * Cin + c0 + c] = tile[c][pp];
}

// -------------------- 1x1 conv as WMMA GEMM (bf16 -> f32) -------------------
// out[b,co,p] = sum_ci W[co,ci] * inbf[b, p, cofs+ci]; inbf pixel-major bf16.
// grid (P/128, Cout/16, B), block 256 (8 waves, one 16x16 tile each).
__global__ void chm_conv1x1_wmma(const __bf16* __restrict__ inbf,
                                 const __bf16* __restrict__ W,
                                 float* __restrict__ out,
                                 int rowW, int cofs, int Cin, int Cout) {
  const int lane = threadIdx.x & 31;
  const int wave = threadIdx.x >> 5;
  const int b = blockIdx.z;
  const int co0 = blockIdx.y << 4;
  const int p = (blockIdx.x << 7) + (wave << 4) + (lane & 15);
  const int kbA = (lane < 16) ? 0 : 8;   // A-fragment K base (16-bit A 16x32 layout)
  const int kbB = (lane < 16) ? 0 : 8;   // B-fragment K base (same K order as A)
  const int mrow = co0 + (lane & 15);
  const __bf16* irow = inbf + ((size_t)b * 65536 + p) * rowW + cofs;
  __builtin_prefetch(irow + 128 * rowW, 0, 1);  // global_prefetch_b8
  f32x8 acc = {0.f, 0.f, 0.f, 0.f, 0.f, 0.f, 0.f, 0.f};
  for (int k0 = 0; k0 < Cin; k0 += 32) {
    const __bf16* wr = W + (size_t)mrow * Cin + k0 + kbA;
    bf16x8 alo = *(const bf16x8*)(wr);
    bf16x8 ahi = *(const bf16x8*)(wr + 16);
    bf16x16 a = __builtin_shufflevector(alo, ahi, 0, 1, 2, 3, 4, 5, 6, 7,
                                        8, 9, 10, 11, 12, 13, 14, 15);
    bf16x8 blo = *(const bf16x8*)(irow + k0 + kbB);
    bf16x8 bhi = *(const bf16x8*)(irow + k0 + kbB + 16);
    bf16x16 bb = __builtin_shufflevector(blo, bhi, 0, 1, 2, 3, 4, 5, 6, 7,
                                         8, 9, 10, 11, 12, 13, 14, 15);
    acc = __builtin_amdgcn_wmma_f32_16x16x32_bf16(false, a, false, bb,
                                                  (short)0, acc, false, false);
  }
  const int ro = (lane < 16) ? 0 : 8;
#pragma unroll
  for (int r = 0; r < 8; ++r)
    out[((size_t)b * Cout + co0 + r + ro) * 65536 + p] = acc[r];
}

// ------------------------- depthwise 3x3 pad 1 ------------------------------
__global__ void chm_dw3x3(const float* __restrict__ in, const float* __restrict__ w,
                          float* __restrict__ out, int Cn) {
  const int p = blockIdx.x * 256 + threadIdx.x;
  const int ch = blockIdx.y, b = blockIdx.z;
  const int y = p >> 8, x = p & 255;
  const float* ip = in + ((size_t)b * Cn + ch) * 65536;
  const float* wp = w + ch * 9;
  float acc = 0.f;
  for (int dy = 0; dy < 3; ++dy) {
    int yy = y + dy - 1;
    if (yy < 0 || yy > 255) continue;
    for (int dx = 0; dx < 3; ++dx) {
      int xx = x + dx - 1;
      if (xx < 0 || xx > 255) continue;
      acc += ip[yy * 256 + xx] * wp[dy * 3 + dx];
    }
  }
  out[((size_t)b * Cn + ch) * 65536 + p] = acc;
}

// --------------------- depthwise 4x4 stride 4 pad 1 -------------------------
__global__ void chm_dw4x4s4(const float* __restrict__ in, const float* __restrict__ w,
                            float* __restrict__ out, int Cn) {
  const int p = blockIdx.x * 256 + threadIdx.x;  // 0..4095
  const int ch = blockIdx.y, b = blockIdx.z;
  const int oy = p >> 6, ox = p & 63;
  const float* ip = in + ((size_t)b * Cn + ch) * 65536;
  const float* wp = w + ch * 16;
  float acc = 0.f;
  for (int ky = 0; ky < 4; ++ky) {
    int yy = oy * 4 - 1 + ky;
    if (yy < 0 || yy > 255) continue;
    for (int kx = 0; kx < 4; ++kx) {
      int xx = ox * 4 - 1 + kx;
      if (xx < 0 || xx > 255) continue;
      acc += ip[yy * 256 + xx] * wp[ky * 4 + kx];
    }
  }
  out[((size_t)b * Cn + ch) * 4096 + p] = acc;
}

// ---- l2norm over d=192 + transpose (b,192,4096)->(b,4096,192) bf16 (+f32) --
__global__ void chm_l2norm_qk(const float* __restrict__ in, __bf16* __restrict__ obf,
                              float* __restrict__ of32) {
  const int lane = threadIdx.x & 31;
  const int wave = threadIdx.x >> 5;
  const int n = blockIdx.x * 8 + wave;
  const int b = blockIdx.y;
  const float* ip = in + (size_t)b * 192 * 4096 + n;
  float vals[6];
  float ss = 0.f;
#pragma unroll
  for (int i = 0; i < 6; ++i) {
    float v = ip[(size_t)(lane + 32 * i) * 4096];
    vals[i] = v; ss += v * v;
  }
  ss = wave_sum(ss);
  float inv = 1.f / fmaxf(sqrtf(ss), 1e-12f);
  size_t base = ((size_t)b * 4096 + n) * 192;
#pragma unroll
  for (int i = 0; i < 6; ++i) {
    int c = lane + 32 * i;
    float v = vals[i] * inv;
    obf[base + c] = f2bf(v);
    if (of32) of32[base + c] = v;
  }
}

// ---- v (b,96,256,256) -> v_s (b,4096,1536): d=(sy*4+sx)*96+c, pix=(sy*64+y2, sx*64+x2)
__global__ void chm_v_rearrange(const float* __restrict__ v, float* __restrict__ vs) {
  size_t i = (size_t)blockIdx.x * 256 + threadIdx.x;
  int d = (int)(i % 1536);
  int n = (int)((i / 1536) % 4096);
  int b = (int)(i / (1536ull * 4096ull));
  int c = d % 96, s = d / 96, sy = s >> 2, sx = s & 3;
  int y2 = n >> 6, x2 = n & 63;
  vs[i] = v[((size_t)b * 96 + c) * 65536 + (size_t)(sy * 64 + y2) * 256 + (sx * 64 + x2)];
}

// ----------------- attention scores S[b,n,m] = qn . kn (WMMA) ---------------
// grid (4096/128, 4096/16, B), block 256.
__global__ void chm_attn_scores(const __bf16* __restrict__ qn,
                                const __bf16* __restrict__ kn,
                                float* __restrict__ S) {
  const int lane = threadIdx.x & 31;
  const int wave = threadIdx.x >> 5;
  const int b = blockIdx.z;
  const int n0 = blockIdx.y << 4;
  const int m0 = (blockIdx.x << 7) + (wave << 4);
  const int col = lane & 15;
  const int kbA = (lane < 16) ? 0 : 8;
  const int kbB = (lane < 16) ? 0 : 8;
  const __bf16* qrow = qn + ((size_t)b * 4096 + n0 + col) * 192;
  const __bf16* krow = kn + ((size_t)b * 4096 + m0 + col) * 192;
  f32x8 acc = {0.f, 0.f, 0.f, 0.f, 0.f, 0.f, 0.f, 0.f};
#pragma unroll
  for (int k0 = 0; k0 < 192; k0 += 32) {
    bf16x8 alo = *(const bf16x8*)(qrow + k0 + kbA);
    bf16x8 ahi = *(const bf16x8*)(qrow + k0 + kbA + 16);
    bf16x16 a = __builtin_shufflevector(alo, ahi, 0, 1, 2, 3, 4, 5, 6, 7,
                                        8, 9, 10, 11, 12, 13, 14, 15);
    bf16x8 blo = *(const bf16x8*)(krow + k0 + kbB);
    bf16x8 bhi = *(const bf16x8*)(krow + k0 + kbB + 16);
    bf16x16 bb = __builtin_shufflevector(blo, bhi, 0, 1, 2, 3, 4, 5, 6, 7,
                                         8, 9, 10, 11, 12, 13, 14, 15);
    acc = __builtin_amdgcn_wmma_f32_16x16x32_bf16(false, a, false, bb,
                                                  (short)0, acc, false, false);
  }
  const int ro = (lane < 16) ? 0 : 8;
#pragma unroll
  for (int r = 0; r < 8; ++r)
    S[((size_t)b * 4096 + n0 + r + ro) * 4096 + m0 + col] = acc[r];
}

// ------- top5 threshold + local mask + softmax + deterministic compaction ---
// one wave per row; grid (4096/8, B), block 256. MAXNZ = 128.
__global__ void chm_attn_mask_softmax(float* __restrict__ S,
                                      const float* __restrict__ temp,
                                      int* __restrict__ nz_cnt,
                                      int* __restrict__ nz_idx,
                                      float* __restrict__ nz_pv) {
  const int lane = threadIdx.x & 31;
  const int wave = threadIdx.x >> 5;
  const int n = blockIdx.x * 8 + wave;
  const int b = blockIdx.y;
  const float tv = temp[0];
  float* Srow = S + ((size_t)b * 4096 + n) * 4096;

  // per-lane top-5 over strided 128 columns
  float t5[5];
#pragma unroll
  for (int j = 0; j < 5; ++j) t5[j] = -INFINITY;
  for (int i = 0; i < 128; ++i) {
    int m = lane + (i << 5);
    float v = Srow[m] * tv;
    if (v > t5[4]) {
      t5[4] = v;
      for (int j = 4; j > 0; --j) {
        if (t5[j] > t5[j - 1]) { float tmp = t5[j]; t5[j] = t5[j - 1]; t5[j - 1] = tmp; }
        else break;
      }
    }
  }
  // 5-round wave merge -> 5th-largest value
  int ptr = 0;
  float kth = -INFINITY;
  for (int r = 0; r < 5; ++r) {
    float cand = (ptr < 5) ? t5[ptr] : -INFINITY;
    float mx = wave_max(cand);
    kth = mx;
    unsigned long long bal = __ballot(cand == mx);
    int leader = (int)__ffsll(bal) - 1;
    if (lane == leader) ptr++;
  }
  const int yn = n >> 6, xn = n & 63;
  // masked row max over nonzeros
  float rm = -INFINITY;
  for (int i = 0; i < 128; ++i) {
    int m = lane + (i << 5);
    float v = Srow[m] * tv;
    int ym = m >> 6, xm = m & 63;
    float w = ((v >= kth) ? 1.f : 0.f) + ((abs(yn - ym) + abs(xn - xm) <= 4) ? 1.f : 0.f);
    float a = v * w;
    if (a != 0.f) rm = fmaxf(rm, a);
  }
  rm = wave_max(rm);
  // exp + sum; store e in-place
  float sum = 0.f;
  for (int i = 0; i < 128; ++i) {
    int m = lane + (i << 5);
    float v = Srow[m] * tv;
    int ym = m >> 6, xm = m & 63;
    float w = ((v >= kth) ? 1.f : 0.f) + ((abs(yn - ym) + abs(xn - xm) <= 4) ? 1.f : 0.f);
    float a = v * w;
    float e = (a != 0.f) ? __expf(a - rm) : 0.f;
    Srow[m] = e;
    sum += e;
  }
  sum = wave_sum(sum);
  float inv = (sum > 0.f) ? (1.f / sum) : 0.f;
  // deterministic compaction: wave-exclusive prefix scan of per-lane counts
  int cl = 0;
  for (int i = 0; i < 128; ++i)
    if (Srow[lane + (i << 5)] > 0.f) cl++;
  int offp = cl;
  for (int d = 1; d < 32; d <<= 1) {
    int t = __shfl_up(offp, d, 32);
    if (lane >= d) offp += t;
  }
  int base = offp - cl;
  int total = __shfl(offp, 31, 32);
  int* idxr = nz_idx + ((size_t)b * 4096 + n) * 128;
  float* pvr = nz_pv + ((size_t)b * 4096 + n) * 128;
  int wi = base;
  for (int i = 0; i < 128; ++i) {
    int m = lane + (i << 5);
    float e = Srow[m];
    if (e > 0.f) {
      if (wi < 128) { idxr[wi] = m; pvr[wi] = e * inv; }
      wi++;
    }
  }
  if (lane == 0) nz_cnt[(size_t)b * 4096 + n] = (total < 128) ? total : 128;
}

// ------------- sparse s@v: out[n,:1536] = sum_m p_m * v_s[m,:] --------------
__global__ void chm_attn_out(const int* __restrict__ nz_cnt,
                             const int* __restrict__ nz_idx,
                             const float* __restrict__ nz_pv,
                             const float* __restrict__ vs,
                             float* __restrict__ so) {
  const int n = blockIdx.x, b = blockIdx.y, t = threadIdx.x;
  const int cnt = nz_cnt[(size_t)b * 4096 + n];
  const int* idxr = nz_idx + ((size_t)b * 4096 + n) * 128;
  const float* pvr = nz_pv + ((size_t)b * 4096 + n) * 128;
  float acc[6] = {0.f, 0.f, 0.f, 0.f, 0.f, 0.f};
  for (int i = 0; i < cnt; ++i) {
    const int m = idxr[i];
    const float p = pvr[i];
    const float* vrow = vs + ((size_t)b * 4096 + m) * 1536 + t;
#pragma unroll
    for (int j = 0; j < 6; ++j) acc[j] += p * vrow[j * 256];
  }
  float* orow = so + ((size_t)b * 4096 + n) * 1536 + t;
#pragma unroll
  for (int j = 0; j < 6; ++j) orow[j * 256] = acc[j];
}

// ---- (b,4096,1536) -> spatial (b,96,256,256) inverse of v_rearrange --------
__global__ void chm_out_rearrange(const float* __restrict__ so, float* __restrict__ osp) {
  size_t i = (size_t)blockIdx.x * 256 + threadIdx.x;
  int p = (int)(i % 65536);
  int c = (int)((i / 65536) % 96);
  int b = (int)(i / (96ull * 65536ull));
  int y = p >> 8, x = p & 255;
  int sy = y >> 6, y2 = y & 63, sx = x >> 6, x2 = x & 63;
  osp[i] = so[((size_t)b * 4096 + (y2 * 64 + x2)) * 1536 + (size_t)((sy * 4 + sx) * 96 + c)];
}

// ---------- l2norm along n=65536 for one (b, channel) row -------------------
__global__ void chm_l2norm_rows(const float* __restrict__ in, float* __restrict__ out,
                                int CinTot, int cofs) {
  __shared__ float red[256];
  const int t = threadIdx.x, c = blockIdx.x, b = blockIdx.y;
  const float* row = in + ((size_t)b * CinTot + cofs + c) * 65536;
  float ss = 0.f;
  for (int j = 0; j < 256; ++j) { float v = row[t + j * 256]; ss += v * v; }
  red[t] = ss;
  __syncthreads();
  for (int s = 128; s > 0; s >>= 1) {
    if (t < s) red[t] += red[t + s];
    __syncthreads();
  }
  float inv = 1.f / fmaxf(sqrtf(red[0]), 1e-12f);
  float* orow = out + ((size_t)b * 96 + c) * 65536;
  for (int j = 0; j < 256; ++j) orow[t + j * 256] = row[t + j * 256] * inv;
}

// ------------------ copy raw v2 channels (192..287) out ---------------------
__global__ void chm_v2copy(const float* __restrict__ qkv, float* __restrict__ v2out) {
  size_t i = (size_t)blockIdx.x * 256 + threadIdx.x;
  int p = (int)(i % 65536);
  int c = (int)((i / 65536) % 96);
  int b = (int)(i / (96ull * 65536ull));
  v2out[i] = qkv[((size_t)b * 288 + 192 + c) * 65536 + p];
}

// ------ a2[b,h,c,d] = softmax_d( temp[h] * sum_n q2n[c,n]*k2n[d,n] ) --------
__global__ void chm_fhr_a2(const float* __restrict__ q2n, const float* __restrict__ k2n,
                           const float* __restrict__ temp, float* __restrict__ a2) {
  __shared__ float av[144];
  const int t = threadIdx.x, head = blockIdx.x, b = blockIdx.y;
  if (t < 144) {
    int cc = t / 12, dd = t % 12;
    const float4* qr = (const float4*)(q2n + ((size_t)b * 96 + head * 12 + cc) * 65536);
    const float4* kr = (const float4*)(k2n + ((size_t)b * 96 + head * 12 + dd) * 65536);
    float acc = 0.f;
    for (int nn = 0; nn < 16384; ++nn) {
      float4 a = qr[nn], k = kr[nn];
      acc += a.x * k.x + a.y * k.y + a.z * k.z + a.w * k.w;
    }
    av[t] = acc * temp[head];
  }
  __syncthreads();
  if (t < 144) {
    int cc = t / 12;
    float mx = -INFINITY;
    for (int d = 0; d < 12; ++d) mx = fmaxf(mx, av[cc * 12 + d]);
    float s = 0.f;
    for (int d = 0; d < 12; ++d) s += __expf(av[cc * 12 + d] - mx);
    a2[(size_t)(b * 8 + head) * 144 + t] = __expf(av[t] - mx) / s;
  }
}

// ---------------- o2pre[b,c,n] = sum_d a2[b,h,c,d] * v2[b,h,d,n] ------------
__global__ void chm_fhr_o2(const float* __restrict__ a2, const float* __restrict__ qkv,
                           float* __restrict__ o2pre) {
  const int p = blockIdx.x * 256 + threadIdx.x;
  const int c = blockIdx.y, b = blockIdx.z;
  const int head = c / 12, cc = c % 12;
  const float* ar = a2 + (size_t)(b * 8 + head) * 144 + cc * 12;
  float acc = 0.f;
#pragma unroll
  for (int d = 0; d < 12; ++d)
    acc += ar[d] * qkv[((size_t)b * 288 + 192 + head * 12 + d) * 65536 + p];
  o2pre[((size_t)b * 96 + c) * 65536 + p] = acc;
}

// ---------------------------------------------------------------------------
extern "C" void kernel_launch(void* const* d_in, const int* in_sizes, int n_in,
                              void* d_out, int out_size, void* d_ws, size_t ws_size,
                              hipStream_t stream) {
  (void)in_sizes; (void)n_in; (void)out_size; (void)ws_size;
  const float* x        = (const float*)d_in[0];
  const float* qk_w     = (const float*)d_in[1];
  const float* qk_dw    = (const float*)d_in[2];
  const float* v_w      = (const float*)d_in[3];
  const float* v_dw     = (const float*)d_in[4];
  const float* k2_w     = (const float*)d_in[5];
  const float* k2_dw    = (const float*)d_in[6];
  const float* q2_w     = (const float*)d_in[7];
  const float* q2_dw    = (const float*)d_in[8];
  const float* proj_w   = (const float*)d_in[9];
  const float* sab_temp = (const float*)d_in[10];
  const float* fqkv_w   = (const float*)d_in[11];
  const float* fqkv_dw  = (const float*)d_in[12];
  const float* fproj_w  = (const float*)d_in[13];
  const float* fhr_temp = (const float*)d_in[14];

  float* outf = (float*)d_out;
  char* ws = (char*)d_ws;

  constexpr size_t P = 65536, N = 4096;
  size_t off = 0;
  auto arena = [&](size_t bytes) { size_t o = off; off += (bytes + 255) & ~size_t(255); return o; };
  // R0: conv mids (<=288ch) and attention score/prob matrix (134MB <= 151MB)
  size_t R0  = arena(288 * P * sizeof(float) * 2);
  size_t R1  = arena(288 * P * sizeof(float) * 2);   // post-dw qk / post-dw qkv
  size_t R2  = arena(96 * P * sizeof(float) * 2);    // v-mid / out_sp / q2n
  size_t R3  = arena(96 * P * sizeof(float) * 2);    // v / sab_out / x2 / o2pre
  size_t Rk  = arena(2 * 192 * N * sizeof(float));   // k downsampled
  size_t Rq  = arena(2 * 192 * N * sizeof(float));   // q downsampled
  size_t Rqb = arena(2 * N * 192 * sizeof(unsigned short));  // qn bf16
  size_t Rkb = arena(2 * N * 192 * sizeof(unsigned short));  // kn bf16
  size_t Rc  = arena(2 * N * sizeof(int));                   // nz counts
  size_t Ri  = arena(2 * N * 128 * sizeof(int));             // nz indices
  size_t Rp  = arena(2 * N * 128 * sizeof(float));           // nz probs
  size_t Ra  = arena(2 * 8 * 144 * sizeof(float));           // a2
  size_t Rx  = arena(2 * P * 96 * sizeof(unsigned short));   // pixel-major bf16, 96ch (reused)
  size_t Rqk = arena(2 * P * 192 * sizeof(unsigned short));  // pixel-major bf16, 192ch
  size_t Rw  = arena(256 * 1024);                            // bf16 weights

  float* pR0 = (float*)(ws + R0);
  float* pR1 = (float*)(ws + R1);
  float* pR2 = (float*)(ws + R2);
  float* pR3 = (float*)(ws + R3);
  float* pRk = (float*)(ws + Rk);
  float* pRq = (float*)(ws + Rq);
  __bf16* pQb = (__bf16*)(ws + Rqb);
  __bf16* pKb = (__bf16*)(ws + Rkb);
  int* pCnt = (int*)(ws + Rc);
  int* pIdx = (int*)(ws + Ri);
  float* pPv = (float*)(ws + Rp);
  float* pA2 = (float*)(ws + Ra);
  __bf16* pXbf  = (__bf16*)(ws + Rx);   // (b,P,96) staging, reused 4x
  __bf16* pQKbf = (__bf16*)(ws + Rqk);  // (b,P,192) staging

  __bf16* wb = (__bf16*)(ws + Rw);
  __bf16* w_qk   = wb;            // 192*96
  __bf16* w_v    = w_qk + 18432;  // 96*96
  __bf16* w_k2   = w_v + 9216;    // 192*96
  __bf16* w_q2   = w_k2 + 18432;  // 192*96
  __bf16* w_proj = w_q2 + 18432;  // 96*96
  __bf16* w_fqkv = w_proj + 9216; // 288*96
  __bf16* w_fprj = w_fqkv + 27648;// 96*96

  // output regions (floats)
  float* o2_out = outf;
  float* ks_out = outf + 12582912;
  float* vs_out = outf + 14155776;
  float* k2_out = outf + 26738688;
  float* v2_out = outf + 39321600;

  const dim3 B256(256);

  // weight conversions (tiny)
  chm_cvt_bf16<<<dim3(72), B256, 0, stream>>>(qk_w, w_qk, 18432);
  chm_cvt_bf16<<<dim3(36), B256, 0, stream>>>(v_w, w_v, 9216);
  chm_cvt_bf16<<<dim3(72), B256, 0, stream>>>(k2_w, w_k2, 18432);
  chm_cvt_bf16<<<dim3(72), B256, 0, stream>>>(q2_w, w_q2, 18432);
  chm_cvt_bf16<<<dim3(36), B256, 0, stream>>>(proj_w, w_proj, 9216);
  chm_cvt_bf16<<<dim3(108), B256, 0, stream>>>(fqkv_w, w_fqkv, 27648);
  chm_cvt_bf16<<<dim3(36), B256, 0, stream>>>(fproj_w, w_fprj, 9216);

  // ---- SAB front-end -------------------------------------------------------
  chm_cvt_tr_bf16<<<dim3(2048, 3, 2), B256, 0, stream>>>(x, pXbf, 96, 0, 96);
  chm_conv1x1_wmma<<<dim3(512, 12, 2), B256, 0, stream>>>(pXbf, w_qk, pR0, 96, 0, 96, 192);
  chm_dw3x3<<<dim3(256, 192, 2), B256, 0, stream>>>(pR0, qk_dw, pR1, 192);   // q=ch0..95 k=96..191
  chm_conv1x1_wmma<<<dim3(512, 6, 2), B256, 0, stream>>>(pXbf, w_v, pR2, 96, 0, 96, 96);
  chm_dw3x3<<<dim3(256, 96, 2), B256, 0, stream>>>(pR2, v_dw, pR3, 96);      // v
  chm_cvt_tr_bf16<<<dim3(2048, 6, 2), B256, 0, stream>>>(pR1, pQKbf, 192, 0, 192);
  chm_conv1x1_wmma<<<dim3(512, 12, 2), B256, 0, stream>>>(pQKbf, w_k2, pR0, 192, 96, 96, 192);
  chm_dw4x4s4<<<dim3(16, 192, 2), B256, 0, stream>>>(pR0, k2_dw, pRk, 192);  // k downsample
  chm_conv1x1_wmma<<<dim3(512, 12, 2), B256, 0, stream>>>(pQKbf, w_q2, pR0, 192, 0, 96, 192);
  chm_dw4x4s4<<<dim3(16, 192, 2), B256, 0, stream>>>(pR0, q2_dw, pRq, 192);  // q downsample

  chm_l2norm_qk<<<dim3(512, 2), B256, 0, stream>>>(pRq, pQb, (float*)nullptr);
  chm_l2norm_qk<<<dim3(512, 2), B256, 0, stream>>>(pRk, pKb, ks_out);        // k_s output
  chm_v_rearrange<<<dim3(49152), B256, 0, stream>>>(pR3, vs_out);            // v_s output

  // ---- sparse attention ----------------------------------------------------
  chm_attn_scores<<<dim3(32, 256, 2), B256, 0, stream>>>(pQb, pKb, pR0);     // S in R0
  chm_attn_mask_softmax<<<dim3(512, 2), B256, 0, stream>>>(pR0, sab_temp, pCnt, pIdx, pPv);
  chm_attn_out<<<dim3(4096, 2), B256, 0, stream>>>(pCnt, pIdx, pPv, vs_out, pR3);  // sab_out in R3
  chm_out_rearrange<<<dim3(49152), B256, 0, stream>>>(pR3, pR2);             // spatial in R2
  chm_cvt_tr_bf16<<<dim3(2048, 3, 2), B256, 0, stream>>>(pR2, pXbf, 96, 0, 96);
  chm_conv1x1_wmma<<<dim3(512, 6, 2), B256, 0, stream>>>(pXbf, w_proj, pR3, 96, 0, 96, 96); // x2

  // ---- FHR -----------------------------------------------------------------
  chm_cvt_tr_bf16<<<dim3(2048, 3, 2), B256, 0, stream>>>(pR3, pXbf, 96, 0, 96);
  chm_conv1x1_wmma<<<dim3(512, 18, 2), B256, 0, stream>>>(pXbf, w_fqkv, pR0, 96, 0, 96, 288);
  chm_dw3x3<<<dim3(256, 288, 2), B256, 0, stream>>>(pR0, fqkv_dw, pR1, 288); // qkv in R1
  chm_l2norm_rows<<<dim3(96, 2), B256, 0, stream>>>(pR1, pR2, 288, 0);       // q2n in R2
  chm_l2norm_rows<<<dim3(96, 2), B256, 0, stream>>>(pR1, k2_out, 288, 96);   // k2 output (normalized)
  chm_v2copy<<<dim3(49152), B256, 0, stream>>>(pR1, v2_out);                 // v2 output (raw)
  chm_fhr_a2<<<dim3(8, 2), dim3(160), 0, stream>>>(pR2, k2_out, fhr_temp, pA2);
  chm_fhr_o2<<<dim3(256, 96, 2), B256, 0, stream>>>(pA2, pR1, pR3);          // o2pre in R3
  chm_cvt_tr_bf16<<<dim3(2048, 3, 2), B256, 0, stream>>>(pR3, pXbf, 96, 0, 96);
  chm_conv1x1_wmma<<<dim3(512, 6, 2), B256, 0, stream>>>(pXbf, w_fprj, o2_out, 96, 0, 96, 96);
}
